// LSHModel_8289286881987
// MI455X (gfx1250) — matrix operand order repaired and checked by reading, not compiled, not run on hardware
//
#include <hip/hip_runtime.h>
#include <hip/hip_bf16.h>

typedef __attribute__((ext_vector_type(16))) __bf16 v16bf;
typedef __attribute__((ext_vector_type(8)))  __bf16 v8bf;
typedef __attribute__((ext_vector_type(8)))  float  v8f;
typedef __attribute__((ext_vector_type(4)))  unsigned int v4u;
typedef __attribute__((ext_vector_type(8)))  int v8i;
typedef __attribute__((ext_vector_type(4)))  int v4i;

#define BATCH 4
#define SEQL  1024
#define EMBD  512
#define NHEAD 8
#define DHEAD 64
#define HIDN  2048
#define NBUCK 64
#define NHYPP 6
#define NCLS  6
#define NENC  6
#define MROWS (BATCH * SEQL)   // 4096

#if defined(__HIP_DEVICE_COMPILE__) && __has_builtin(__builtin_amdgcn_tensor_load_to_lds)
#define HAVE_TDM 1
#else
#define HAVE_TDM 0
#endif

__device__ __forceinline__ v8f vzero8() {
  v8f z = {0.f, 0.f, 0.f, 0.f, 0.f, 0.f, 0.f, 0.f};
  return z;
}

__device__ __forceinline__ v16bf concat8(v8bf lo, v8bf hi) {
  return __builtin_shufflevector(lo, hi, 0, 1, 2, 3, 4, 5, 6, 7,
                                 8, 9, 10, 11, 12, 13, 14, 15);
}

__device__ __forceinline__ v8bf ld8(const __bf16* p) {  // 16B vector load
  return *(const v8bf*)p;
}

__device__ __forceinline__ v8f wmma_bf16(v16bf a, v16bf b, v8f c) {
  // D = A(16x32 bf16) x B(32x16 bf16) + C(16x16 f32)
  return __builtin_amdgcn_wmma_f32_16x16x32_bf16(
      /*neg_a=*/false, a, /*neg_b=*/false, b,
      /*c_mod=*/(short)0, c, /*reuse_a=*/false, /*reuse_b=*/false);
}

#if HAVE_TDM
// Issue a TDM load of a [128 x 32] bf16 tile (row-major) from a [Mdim x Kdim]
// bf16 tensor into LDS at lds_off.  D# per CDNA5 ISA 8.3-8.5.
__device__ __forceinline__ void tdm_load_tile_a(const __bf16* gaddr, unsigned lds_off,
                                                int Kdim, int Mdim) {
  unsigned long long ga = (unsigned long long)(size_t)gaddr;
  v4u g0;
  g0[0] = 1u;                                           // count=1, user D#
  g0[1] = lds_off;                                      // LDS byte address
  g0[2] = (unsigned)(ga & 0xffffffffu);                 // global_addr[31:0]
  g0[3] = (unsigned)((ga >> 32) & 0x01ffffffu) | (2u << 30);  // [56:32], type=2
  v8i g1;
  g1[0] = 0x00010000;                                   // data_size=1 (2B), mask=0
  g1[1] = (int)(((unsigned)Kdim & 0xffffu) << 16);      // tensor_dim0[15:0]
  g1[2] = (int)((((unsigned)Kdim >> 16) & 0xffffu) |    // tensor_dim0[31:16]
                (((unsigned)Mdim & 0xffffu) << 16));    // tensor_dim1[15:0]
  g1[3] = (int)((((unsigned)Mdim >> 16) & 0xffffu) |    // tensor_dim1[31:16]
                (32u << 16));                           // tile_dim0 = 32
  g1[4] = 128;                                          // tile_dim1=128, tile_dim2=0
  g1[5] = Kdim;                                         // tensor_dim0_stride[31:0]
  g1[6] = 0;
  g1[7] = 0;
  v4i gz = {0, 0, 0, 0};
#if __clang_major__ >= 23
  v8i gz8 = {0, 0, 0, 0, 0, 0, 0, 0};
  __builtin_amdgcn_tensor_load_to_lds(g0, g1, gz, gz, gz8, 0);
#else
  __builtin_amdgcn_tensor_load_to_lds(g0, g1, gz, gz, 0);
#endif
}
#endif

// ---------------------------------------------------------------------------
// f32 -> bf16 weight conversion
// ---------------------------------------------------------------------------
__global__ void k_f32_to_bf16(const float* __restrict__ s, __bf16* __restrict__ d, int n) {
  int i = blockIdx.x * blockDim.x + threadIdx.x;
  int stride = gridDim.x * blockDim.x;
  for (; i < n; i += stride) d[i] = (__bf16)s[i];
}

// ---------------------------------------------------------------------------
// Embedding lookup + sinusoidal positional encoding -> f32 and bf16 copies
// ---------------------------------------------------------------------------
__global__ void k_embed_pos(const int* __restrict__ tok, const float* __restrict__ emb,
                            float* __restrict__ xf, __bf16* __restrict__ xb) {
  int bn = blockIdx.x;               // 0..MROWS-1
  int n = bn & (SEQL - 1);
  int t = tok[bn];
  const float kinv = 9.210340371976184f / (float)EMBD;  // ln(10000)/E
  for (int c = threadIdx.x; c < EMBD; c += blockDim.x) {
    int i2 = c & ~1;
    float div = __expf(-(float)i2 * kinv);
    float ang = (float)n * div;
    float pe = (c & 1) ? __cosf(ang) : __sinf(ang);
    float v = emb[(size_t)t * EMBD + c] + pe;
    xf[(size_t)bn * EMBD + c] = v;
    xb[(size_t)bn * EMBD + c] = (__bf16)v;
  }
}

// ---------------------------------------------------------------------------
// bf16 WMMA GEMM: C[M,N] = act(A[M,K] @ B[K,N] + bias) (+ resid)
// workgroup tile 128x64, 8 waves, K-step 32.
// A tile: double-buffered LDS, filled by the Tensor Data Mover (TDM) so the
// DMA of tile k+1 overlaps the B-load + WMMA of tile k (TENSORcnt pipeline).
// B tile: stored in LDS in fragment-major layout [ntile][lane][16] so
// B-fragment reads are 2x ds_load_b128.
// ---------------------------------------------------------------------------
__global__ __launch_bounds__(256) void k_gemm_wmma(
    const __bf16* __restrict__ A, const __bf16* __restrict__ B,
    const float* __restrict__ bias, const float* __restrict__ resid,
    float* __restrict__ Cf, __bf16* __restrict__ Cb,
    int M, int N, int K, int relu) {
  __shared__ alignas(16) __bf16 As[2][128 * 32];
  __shared__ alignas(16) __bf16 Bs[4 * 32 * 16];
  const int mbase = blockIdx.y * 128;
  const int nbase = blockIdx.x * 64;
  const int tid = threadIdx.x;
  const int wave = tid >> 5, lane = tid & 31;
  const int lh = lane & 15, sel = lane >> 4;

  v8f acc[4];
#pragma unroll
  for (int t = 0; t < 4; ++t) acc[t] = vzero8();

  const int nsteps = K / 32;
  const __bf16* Abase = A + (size_t)mbase * K;

#if HAVE_TDM
  // prologue: DMA first A tile
  if (wave == 0) {
    tdm_load_tile_a(Abase, (unsigned)(size_t)(const void*)&As[0][0], K, M);
    __builtin_amdgcn_s_wait_tensorcnt(0);
  }
  __syncthreads();
#endif

  for (int step = 0; step < nsteps; ++step) {
    const int k0 = step * 32;
    const int cur = step & 1;

#if HAVE_TDM
    // kick the DMA for the next A tile into the other buffer (async)
    if (wave == 0 && step + 1 < nsteps)
      tdm_load_tile_a(Abase + (k0 + 32),
                      (unsigned)(size_t)(const void*)&As[1 - cur][0], K, M);
#else
    // cooperative A tile load: two 16B vector loads + stores per thread
    {
      const int ar = tid >> 1;           // row 0..127
      const int ac = (tid & 1) * 16;     // col 0 or 16
      if (k0 + 32 < K)
        __builtin_prefetch(&Abase[(size_t)ar * K + k0 + 32], 0, 1);
      const __bf16* ap = Abase + (size_t)ar * K + k0 + ac;
      *(v8bf*)&As[cur][ar * 32 + ac] = ld8(ap);
      *(v8bf*)&As[cur][ar * 32 + ac + 8] = ld8(ap + 8);
    }
#endif

    // B tile 32x64: coalesced 16B global load, swizzled scatter to LDS
    {
      int flat = tid * 8;                // 2048 elements
      int kr = flat >> 6;                // 0..31  (k row)
      int cb = flat & 63;                // col base, multiple of 8
      if (k0 + 32 < K)
        __builtin_prefetch(&B[(size_t)(k0 + 32 + kr) * N + nbase + cb], 0, 1);
      v8bf gv = ld8(B + (size_t)(k0 + kr) * N + nbase + cb);
      int e = kr & 15;
      int hl = kr & 16;                  // lane-half select
#pragma unroll
      for (int j = 0; j < 8; ++j) {
        int cj = cb + j;
        int nt = cj >> 4, nl = cj & 15;
        Bs[(nt * 32 + nl + hl) * 16 + e] = gv[j];
      }
    }
    __syncthreads();

    // A fragment: 2x contiguous 16B LDS loads
    int ml = wave * 16 + lh;
    int akb = sel * 16;
    const __bf16* ap = &As[cur][ml * 32 + akb];
    v16bf af = concat8(*(const v8bf*)ap, *(const v8bf*)(ap + 8));
#pragma unroll
    for (int nt = 0; nt < 4; ++nt) {
      const __bf16* bp = &Bs[(nt * 32 + lane) * 16];
      v16bf bfr = concat8(*(const v8bf*)bp, *(const v8bf*)(bp + 8));
      acc[nt] = wmma_bf16(af, bfr, acc[nt]);
    }

#if HAVE_TDM
    // next A tile must have landed before anyone touches it
    if (wave == 0 && step + 1 < nsteps) __builtin_amdgcn_s_wait_tensorcnt(0);
#endif
    __syncthreads();
  }

#pragma unroll
  for (int nt = 0; nt < 4; ++nt) {
#pragma unroll
    for (int r = 0; r < 8; ++r) {
      int mg = mbase + wave * 16 + r + sel * 8;
      int ng = nbase + nt * 16 + lh;
      float v = acc[nt][r];
      if (bias) v += bias[ng];
      if (relu) v = fmaxf(v, 0.f);
      if (resid) v += resid[(size_t)mg * N + ng];
      if (Cf) Cf[(size_t)mg * N + ng] = v;
      if (Cb) Cb[(size_t)mg * N + ng] = (__bf16)v;
    }
  }
}

// ---------------------------------------------------------------------------
// Pack q into [b,h,n,dh] bf16, v TRANSPOSED into [b,h,dh,n] bf16 (so P@V
// B-fragments are contiguous), and compute LSH buckets.
// ---------------------------------------------------------------------------
__global__ void k_pack_qv_buckets(const float* __restrict__ q, const float* __restrict__ v,
                                  const float* __restrict__ hyper,
                                  __bf16* __restrict__ qk, __bf16* __restrict__ vkT,
                                  int* __restrict__ buckets) {
  int id = blockIdx.x * blockDim.x + threadIdx.x;  // (b*8+h)*1024+n
  if (id >= BATCH * NHEAD * SEQL) return;
  int n = id & (SEQL - 1);
  int h = (id >> 10) & (NHEAD - 1);
  int b = id >> 13;
  size_t src = ((size_t)(b * SEQL + n)) * EMBD + h * DHEAD;
  size_t qdst = (size_t)id * DHEAD;
  size_t vbase = (size_t)(b * NHEAD + h) * DHEAD * SEQL;
  float proj[NHYPP];
#pragma unroll
  for (int y = 0; y < NHYPP; ++y) proj[y] = hyper[DHEAD * NHYPP + y];  // bias row
  for (int d = 0; d < DHEAD; ++d) {
    float qv = q[src + d];
    qk[qdst + d] = (__bf16)qv;
    vkT[vbase + (size_t)d * SEQL + n] = (__bf16)v[src + d];
#pragma unroll
    for (int y = 0; y < NHYPP; ++y) proj[y] += qv * hyper[d * NHYPP + y];
  }
  int bu = 0;
#pragma unroll
  for (int y = 0; y < NHYPP; ++y) bu |= (proj[y] >= 0.f) ? (1 << y) : 0;
  buckets[id] = bu;
}

// ---------------------------------------------------------------------------
// LSH attention, flash-style two-pass, WMMA for S=QQ^T and O=PV
// one block = (b, h, 16-query tile), 4 waves
// ---------------------------------------------------------------------------
__global__ __launch_bounds__(128) void k_lsh_attn(
    const __bf16* __restrict__ qk, const __bf16* __restrict__ vkT,
    const int* __restrict__ buckets, float* __restrict__ out) {
  __shared__ alignas(16) __bf16 Pt[4 * 16 * 32];  // per-wave exp(S) staging
  __shared__ float wred[4][16];
  __shared__ float rowmax[16];
  __shared__ float rowsum[16];
  __shared__ int qb[16];

  int bid = blockIdx.x;              // B*H*(SEQ/16)
  int qt = bid & 63;
  int h = (bid >> 6) & 7;
  int b = bid >> 9;
  int tid = threadIdx.x;
  int wave = tid >> 5, lane = tid & 31;
  int lh = lane & 15, sel = lane >> 4;

  const __bf16* qh = qk + ((size_t)(b * NHEAD + h)) * SEQL * DHEAD;
  const __bf16* vT = vkT + ((size_t)(b * NHEAD + h)) * DHEAD * SEQL;
  const int* bh = buckets + (size_t)(b * NHEAD + h) * SEQL;
  const float scale = 0.044194173824159216f;  // 1/sqrt(512)

  if (tid < 16) qb[tid] = bh[qt * 16 + tid];
  __syncthreads();

  // Query A-fragments for the two K=32 halves of dh: 2x 16B loads each
  int akb = sel * 16;
  int bkb = sel * 16;
  v16bf aq[2];
  {
    const __bf16* qp = qh + (size_t)(qt * 16 + lh) * DHEAD;
#pragma unroll
    for (int ks = 0; ks < 2; ++ks)
      aq[ks] = concat8(ld8(qp + ks * 32 + akb), ld8(qp + ks * 32 + akb + 8));
  }

  // ---- pass 1: row maxima of scaled, bucket-counted scores ----
  float pm[8];
#pragma unroll
  for (int r = 0; r < 8; ++r) pm[r] = -3.0e38f;

  for (int i = 0; i < 16; ++i) {
    int kc = wave + i * 4;           // 64 key chunks of 16, split over waves
    int key0 = kc * 16;
    const __bf16* kp = qh + (size_t)(key0 + lh) * DHEAD;
    v8f s = vzero8();
#pragma unroll
    for (int ks = 0; ks < 2; ++ks) {
      v16bf bfr = concat8(ld8(kp + ks * 32 + bkb), ld8(kp + ks * 32 + bkb + 8));
      s = wmma_bf16(aq[ks], bfr, s);
    }
    int kb = bh[key0 + lh];
#pragma unroll
    for (int r = 0; r < 8; ++r) {
      int mrow = r + sel * 8;
      float cnt = (float)(NBUCK - 2 + ((qb[mrow] == kb) ? 1 : 0));
      pm[r] = fmaxf(pm[r], s[r] * scale * cnt);
    }
  }
#pragma unroll
  for (int m = 1; m < 16; m <<= 1) {
#pragma unroll
    for (int r = 0; r < 8; ++r) pm[r] = fmaxf(pm[r], __shfl_xor(pm[r], m, 32));
  }
  if (lane == 0) {
#pragma unroll
    for (int r = 0; r < 8; ++r) wred[wave][r] = pm[r];
  }
  if (lane == 16) {
#pragma unroll
    for (int r = 0; r < 8; ++r) wred[wave][8 + r] = pm[r];
  }
  __syncthreads();
  if (tid < 16) {
    float mx = wred[0][tid];
    for (int w = 1; w < 4; ++w) mx = fmaxf(mx, wred[w][tid]);
    rowmax[tid] = mx;
  }
  __syncthreads();

  // ---- pass 2: recompute S, exponentiate, O += P @ V (wave owns 16 dh cols) ----
  v8f o = vzero8();
  float ps[8];
#pragma unroll
  for (int r = 0; r < 8; ++r) ps[r] = 0.f;
  int dh = wave * 16 + lh;
  const __bf16* vrow = vT + (size_t)dh * SEQL;

  for (int kc = 0; kc < 32; ++kc) {  // 32-key chunks
#pragma unroll
    for (int sub = 0; sub < 2; ++sub) {
      int key0 = kc * 32 + sub * 16;
      const __bf16* kp = qh + (size_t)(key0 + lh) * DHEAD;
      v8f s = vzero8();
#pragma unroll
      for (int ks = 0; ks < 2; ++ks) {
        v16bf bfr = concat8(ld8(kp + ks * 32 + bkb), ld8(kp + ks * 32 + bkb + 8));
        s = wmma_bf16(aq[ks], bfr, s);
      }
      int kb = bh[key0 + lh];
#pragma unroll
      for (int r = 0; r < 8; ++r) {
        int mrow = r + sel * 8;
        float cnt = (float)(NBUCK - 2 + ((qb[mrow] == kb) ? 1 : 0));
        float p = __expf(s[r] * scale * cnt - rowmax[mrow]);
        Pt[(wave * 16 + mrow) * 32 + sub * 16 + lh] = (__bf16)p;  // C->A transpose
        ps[r] += p;
      }
    }
    // A fragment of P from this wave's private staging: 2x contiguous b128
    const __bf16* pp = &Pt[(wave * 16 + lh) * 32 + akb];
    v16bf af = concat8(*(const v8bf*)pp, *(const v8bf*)(pp + 8));
    // B fragment of V (transposed layout): contiguous in keys
    const __bf16* vp = vrow + kc * 32 + bkb;
    v16bf bfr = concat8(ld8(vp), ld8(vp + 8));
    o = wmma_bf16(af, bfr, o);
  }

  // row sums (identical across waves; take wave 0's)
#pragma unroll
  for (int m = 1; m < 16; m <<= 1) {
#pragma unroll
    for (int r = 0; r < 8; ++r) ps[r] += __shfl_xor(ps[r], m, 32);
  }
  if (wave == 0 && lane == 0) {
#pragma unroll
    for (int r = 0; r < 8; ++r) rowsum[r] = ps[r];
  }
  if (wave == 0 && lane == 16) {
#pragma unroll
    for (int r = 0; r < 8; ++r) rowsum[8 + r] = ps[r];
  }
  __syncthreads();

#pragma unroll
  for (int r = 0; r < 8; ++r) {
    int mrow = r + sel * 8;
    int mg = qt * 16 + mrow;
    float val = o[r] / rowsum[mrow];
    out[(size_t)(b * SEQL + mg) * EMBD + h * DHEAD + dh] = val;
  }
}

// ---------------------------------------------------------------------------
// LayerNorm over last dim (512), one block per row
// ---------------------------------------------------------------------------
__global__ __launch_bounds__(256) void k_layernorm(
    const float* __restrict__ in, const float* __restrict__ g, const float* __restrict__ be,
    float* __restrict__ outf, __bf16* __restrict__ outb) {
  __shared__ float red[256];
  int row = blockIdx.x;
  const float* x = in + (size_t)row * EMBD;
  int c0 = threadIdx.x, c1 = threadIdx.x + 256;
  float a = x[c0], c = x[c1];
  red[threadIdx.x] = a + c;
  __syncthreads();
  for (int s = 128; s > 0; s >>= 1) {
    if (threadIdx.x < s) red[threadIdx.x] += red[threadIdx.x + s];
    __syncthreads();
  }
  float mean = red[0] / (float)EMBD;
  __syncthreads();
  float da = a - mean, dc = c - mean;
  red[threadIdx.x] = da * da + dc * dc;
  __syncthreads();
  for (int s = 128; s > 0; s >>= 1) {
    if (threadIdx.x < s) red[threadIdx.x] += red[threadIdx.x + s];
    __syncthreads();
  }
  float inv = rsqrtf(red[0] / (float)EMBD + 1e-5f);
  float y0 = da * inv * g[c0] + be[c0];
  float y1 = dc * inv * g[c1] + be[c1];
  outf[(size_t)row * EMBD + c0] = y0;
  outf[(size_t)row * EMBD + c1] = y1;
  outb[(size_t)row * EMBD + c0] = (__bf16)y0;
  outb[(size_t)row * EMBD + c1] = (__bf16)y1;
}

// ---------------------------------------------------------------------------
// Final classifier: out[b,c] = x_flat[b,:] . Wm[:,c] + bm[c]
// ---------------------------------------------------------------------------
__global__ __launch_bounds__(256) void k_classifier(
    const float* __restrict__ x, const float* __restrict__ Wm,
    const float* __restrict__ bm, float* __restrict__ out) {
  __shared__ float red[256];
  int b = blockIdx.x / NCLS, c = blockIdx.x % NCLS;
  const float* xb = x + (size_t)b * (EMBD * SEQL);
  float s = 0.f;
  for (int i = threadIdx.x; i < EMBD * SEQL; i += 256)
    s += xb[i] * Wm[(size_t)i * NCLS + c];
  red[threadIdx.x] = s;
  __syncthreads();
  for (int t = 128; t > 0; t >>= 1) {
    if (threadIdx.x < t) red[threadIdx.x] += red[threadIdx.x + t];
    __syncthreads();
  }
  if (threadIdx.x == 0) out[b * NCLS + c] = red[0] + bm[c];
}

// ---------------------------------------------------------------------------
extern "C" void kernel_launch(void* const* d_in, const int* in_sizes, int n_in,
                              void* d_out, int out_size, void* d_ws, size_t ws_size,
                              hipStream_t stream) {
  (void)in_sizes; (void)n_in; (void)out_size; (void)ws_size;
  const int*   inputs = (const int*)  d_in[0];
  const float* emb    = (const float*)d_in[1];
  const float* gamma  = (const float*)d_in[2];
  const float* beta   = (const float*)d_in[3];
  const float* Wq     = (const float*)d_in[4];
  const float* bq     = (const float*)d_in[5];
  const float* Wv     = (const float*)d_in[6];
  const float* bv     = (const float*)d_in[7];
  const float* hyper  = (const float*)d_in[8];
  const float* W1     = (const float*)d_in[9];
  const float* b1     = (const float*)d_in[10];
  const float* W2     = (const float*)d_in[11];
  const float* b2     = (const float*)d_in[12];
  const float* Wm     = (const float*)d_in[13];
  const float* bm     = (const float*)d_in[14];
  float* out = (float*)d_out;

  char* w = (char*)d_ws;
  size_t off = 0;
  auto take = [&](size_t bytes) -> char* {
    char* p = w + off;
    off = (off + bytes + 255) & ~(size_t)255;
    return p;
  };
  float*  xf   = (float*) take((size_t)MROWS * EMBD * 4);
  __bf16* xb   = (__bf16*)take((size_t)MROWS * EMBD * 2);
  float*  qf   = (float*) take((size_t)MROWS * EMBD * 4);
  float*  vf   = (float*) take((size_t)MROWS * EMBD * 4);
  __bf16* qkb  = (__bf16*)take((size_t)MROWS * EMBD * 2);
  __bf16* vkT  = (__bf16*)take((size_t)MROWS * EMBD * 2);
  int*    bks  = (int*)   take((size_t)BATCH * NHEAD * SEQL * 4);
  float*  tmpf = (float*) take((size_t)MROWS * EMBD * 4);
  __bf16* h1b  = (__bf16*)take((size_t)MROWS * HIDN * 2);
  __bf16* wqb  = (__bf16*)take((size_t)EMBD * EMBD * 2);
  __bf16* wvb  = (__bf16*)take((size_t)EMBD * EMBD * 2);
  __bf16* w1b  = (__bf16*)take((size_t)EMBD * HIDN * 2);
  __bf16* w2b  = (__bf16*)take((size_t)HIDN * EMBD * 2);

  k_f32_to_bf16<<<256, 256, 0, stream>>>(Wq, wqb, EMBD * EMBD);
  k_f32_to_bf16<<<256, 256, 0, stream>>>(Wv, wvb, EMBD * EMBD);
  k_f32_to_bf16<<<1024, 256, 0, stream>>>(W1, w1b, EMBD * HIDN);
  k_f32_to_bf16<<<1024, 256, 0, stream>>>(W2, w2b, HIDN * EMBD);
  k_embed_pos<<<MROWS, 256, 0, stream>>>(inputs, emb, xf, xb);

  for (int it = 0; it < NENC; ++it) {
    dim3 g_qv(EMBD / 64, MROWS / 128);
    k_gemm_wmma<<<g_qv, 256, 0, stream>>>(xb, wqb, bq, nullptr, qf, nullptr,
                                          MROWS, EMBD, EMBD, 0);
    k_gemm_wmma<<<g_qv, 256, 0, stream>>>(xb, wvb, bv, nullptr, vf, nullptr,
                                          MROWS, EMBD, EMBD, 0);
    k_pack_qv_buckets<<<(BATCH * NHEAD * SEQL) / 256, 256, 0, stream>>>(
        qf, vf, hyper, qkb, vkT, bks);
    k_lsh_attn<<<BATCH * NHEAD * (SEQL / 16), 128, 0, stream>>>(qkb, vkT, bks, tmpf);
    k_layernorm<<<MROWS, 256, 0, stream>>>(tmpf, gamma, beta, xf, xb);
    dim3 g_h1(HIDN / 64, MROWS / 128);
    k_gemm_wmma<<<g_h1, 256, 0, stream>>>(xb, w1b, b1, nullptr, nullptr, h1b,
                                          MROWS, HIDN, EMBD, 1);
    dim3 g_h2(EMBD / 64, MROWS / 128);
    k_gemm_wmma<<<g_h2, 256, 0, stream>>>(h1b, w2b, b2, xf, tmpf, nullptr,
                                          MROWS, EMBD, HIDN, 0);
    k_layernorm<<<MROWS, 256, 0, stream>>>(tmpf, gamma, beta, xf, xb);
  }
  k_classifier<<<BATCH * NCLS, 256, 0, stream>>>(xf, Wm, bm, out);
}
